// MsTNL_52261162058074
// MI455X (gfx1250) — matrix operand order repaired and checked by reading, not compiled, not run on hardware
//
#include <hip/hip_runtime.h>

// ---------------------------------------------------------------------------
// MsTNL for MI455X (gfx1250): bf16 WMMA GEMM core with async global->LDS
// staging (double-buffered), plus streaming pre/post passes.
// ---------------------------------------------------------------------------

typedef __bf16 bf16_t;
typedef __attribute__((ext_vector_type(16))) __bf16 v16bf;
typedef __attribute__((ext_vector_type(8)))  float  v8f;
typedef __attribute__((ext_vector_type(4)))  unsigned int u32x4;
typedef int v4i __attribute__((vector_size(4 * sizeof(int))));  // matches builtin param

#define F_BIASM    1
#define F_BIASN    2
#define F_ADDEND   4
#define F_RELU     8
#define F_SCALEM   16
#define F_ACCUM    32
#define F_GAMMARES 64
#define F_OUTF     128
#define F_OUTH     256

// ---- CDNA5 async global->LDS copy (ASYNCcnt-tracked), with sync fallback ----
#if __has_builtin(__builtin_amdgcn_global_load_async_to_lds_b128)
#define HAVE_ASYNC_COPY 1
#endif

__device__ __forceinline__ void copy16_g2l(const bf16_t* src, bf16_t* dst) {
#ifdef HAVE_ASYNC_COPY
  __builtin_amdgcn_global_load_async_to_lds_b128((v4i*)src, (v4i*)dst, 0, 0);
#else
  *(u32x4*)dst = *(const u32x4*)src;
#endif
}

template <int N>
__device__ __forceinline__ void wait_async_imm() {
#ifdef HAVE_ASYNC_COPY
#if __has_builtin(__builtin_amdgcn_s_wait_asynccnt)
  __builtin_amdgcn_s_wait_asynccnt((unsigned short)N);
#else
  if constexpr (N == 0)      asm volatile("s_wait_asynccnt 0x0" ::: "memory");
  else if constexpr (N == 2) asm volatile("s_wait_asynccnt 0x2" ::: "memory");
  else                       asm volatile("s_wait_asynccnt 0x4" ::: "memory");
#endif
#endif
}

struct GemmArgs {
  const bf16_t* A; long long sAm, sAk, aStrideB;
  const bf16_t* B; long long sBn, sBk, bStrideB;
  float* outF; bf16_t* outH; long long sCm, cStrideB;
  const float* biasM; const float* biasN; const float* scaleM;
  const float* addend; const float* residual; const float* gammaPtr;
  long long sDm, dStrideB, sRm, rStrideB;
  int M, N, K, flags;
};

__device__ __forceinline__ v8f wmma_bf16(v16bf a, v16bf b, v8f c) {
  return __builtin_amdgcn_wmma_f32_16x16x32_bf16(false, a, false, b, (short)0, c,
                                                 false, false);
}

// Stage one 64(row) x 32(k) bf16 tile into LDS.
// MODE 0: k contiguous (sK==1)  -> 16B vector copies (async when available).
// MODE 1: m contiguous (sM==1)  -> 16B vector load along m + LDS scatter.
// PERM: apply the A-fragment K-octet permutation {0-7,16-23,8-15,24-31} so a
// lane's 16-element fragment is one contiguous 32B LDS read (ISA 7.12.2).
template <int MODE, bool PERM>
__device__ __forceinline__ void stage_tile(const bf16_t* __restrict__ base,
                                           long long sM, long long sK,
                                           int rc0, int k0, bf16_t* sh, int t) {
  if (MODE == 0) {
#pragma unroll
    for (int c = t; c < 256; c += 128) {
      const int row = c >> 2;
      const int kk  = (c & 3) * 8;
      int pos = kk;
      if (PERM) { const int swp = ((kk >> 3) ^ (kk >> 4)) & 1; pos = swp ? (kk ^ 24) : kk; }
      const bf16_t* src = base + (long long)(rc0 + row) * sM + (k0 + kk);
      copy16_g2l(src, sh + row * 32 + pos);
    }
  } else {
#pragma unroll
    for (int c = t; c < 256; c += 128) {
      const int k  = c >> 3;         // 0..31
      const int mg = (c & 7) * 8;    // 0,8,...,56
      int pos = k;
      if (PERM) { const int swp = ((k >> 3) ^ (k >> 4)) & 1; pos = swp ? (k ^ 24) : k; }
      const bf16_t* src = base + (rc0 + mg) + (long long)(k0 + k) * sK;  // sM == 1
      const u32x4 vv = *(const u32x4*)src;
      bf16_t tmp[8];
      *(u32x4*)tmp = vv;
#pragma unroll
      for (int j = 0; j < 8; ++j) sh[(mg + j) * 32 + pos] = tmp[j];
    }
  }
}

// Generic GEMM: C[M,N] = A[M,K] * B^T[N,K]. 128 threads (4 waves),
// block tile 64x64, wave tile 32x32 (4 x v_wmma_f32_16x16x32_bf16 per k-step),
// double-buffered LDS staging overlapped with compute via ASYNCcnt.
template <int AM, int BM>
__global__ __launch_bounds__(128) void gemm_bf16_kernel(GemmArgs g) {
  __shared__ __attribute__((aligned(32))) bf16_t sA[2][64 * 32];
  __shared__ __attribute__((aligned(32))) bf16_t sB[2][64 * 32];

  const int t    = threadIdx.x;
  const int lane = t & 31;
  const int wave = t >> 5;
  const int wm   = wave >> 1;
  const int wn   = wave & 1;
  const int m0   = blockIdx.y * 64;
  const int n0   = blockIdx.x * 64;
  const long long bz = blockIdx.z;

  const bf16_t* Ab = g.A + bz * g.aStrideB;
  const bf16_t* Bb = g.B + bz * g.bStrideB;

  constexpr int NASYNC = ((AM == 0) ? 2 : 0) + ((BM == 0) ? 2 : 0);

  v8f acc[2][2];
#pragma unroll
  for (int i = 0; i < 2; ++i)
#pragma unroll
    for (int j = 0; j < 2; ++j)
#pragma unroll
      for (int v = 0; v < 8; ++v) acc[i][j][v] = 0.0f;

  const int nsteps = g.K >> 5;
  stage_tile<AM, true >(Ab, g.sAm, g.sAk, m0, 0, &sA[0][0], t);
  stage_tile<BM, true >(Bb, g.sBn, g.sBk, n0, 0, &sB[0][0], t);

  for (int s = 0; s < nsteps; ++s) {
    const int cur = s & 1;
    if (s + 1 < nsteps) {
      stage_tile<AM, true >(Ab, g.sAm, g.sAk, m0, (s + 1) * 32, &sA[cur ^ 1][0], t);
      stage_tile<BM, true >(Bb, g.sBn, g.sBk, n0, (s + 1) * 32, &sB[cur ^ 1][0], t);
      wait_async_imm<NASYNC>();   // own copies for the *current* tile are done
    } else {
      wait_async_imm<0>();
    }
    __syncthreads();              // everyone's copies for current tile are done

    const int r = lane & 15;
    const int h = lane >> 4;
    const bf16_t* cA = &sA[cur][0];
    const bf16_t* cB = &sB[cur][0];
    v16bf a0 = *(const v16bf*)&cA[(wm * 32 + r) * 32 + h * 16];
    v16bf a1 = *(const v16bf*)&cA[(wm * 32 + 16 + r) * 32 + h * 16];
    v16bf b0 = *(const v16bf*)&cB[(wn * 32 + r) * 32 + h * 16];
    v16bf b1 = *(const v16bf*)&cB[(wn * 32 + 16 + r) * 32 + h * 16];
    acc[0][0] = wmma_bf16(a0, b0, acc[0][0]);
    acc[0][1] = wmma_bf16(a0, b1, acc[0][1]);
    acc[1][0] = wmma_bf16(a1, b0, acc[1][0]);
    acc[1][1] = wmma_bf16(a1, b1, acc[1][1]);
    __syncthreads();              // compute done before next tile overwrites LDS
  }

  // ---- epilogue: C layout lane(col)=lane&15, row = vgpr + (lane>=16 ? 8 : 0)
  const int r = lane & 15;
  const int h = lane >> 4;
  const int flags = g.flags;
#pragma unroll
  for (int i = 0; i < 2; ++i) {
#pragma unroll
    for (int j = 0; j < 2; ++j) {
      const int n = n0 + wn * 32 + j * 16 + r;
#pragma unroll
      for (int v = 0; v < 8; ++v) {
        const int m = m0 + wm * 32 + i * 16 + h * 8 + v;
        float val = acc[i][j][v];
        if (flags & F_SCALEM)  val *= g.scaleM[m];
        if (flags & F_BIASM)   val += g.biasM[m];
        if (flags & F_BIASN)   val += g.biasN[n];
        if (flags & F_ADDEND)  val += g.addend[bz * g.dStrideB + (long long)m * g.sDm + n];
        if (flags & F_RELU)    val = fmaxf(val, 0.0f);
        if (flags & F_GAMMARES)
          val = g.gammaPtr[0] * val + g.residual[bz * g.rStrideB + (long long)m * g.sRm + n];
        const long long ci = bz * g.cStrideB + (long long)m * g.sCm + n;
        if (flags & F_OUTF) {
          if (flags & F_ACCUM) g.outF[ci] += val; else g.outF[ci] = val;
        }
        if (flags & F_OUTH) g.outH[ci] = (bf16_t)val;
      }
    }
  }
}

static void launch_gemm(const GemmArgs& g, dim3 grid, int amode, int bmode,
                        hipStream_t stream) {
  if (amode == 0 && bmode == 0)      gemm_bf16_kernel<0, 0><<<grid, 128, 0, stream>>>(g);
  else if (amode == 1 && bmode == 0) gemm_bf16_kernel<1, 0><<<grid, 128, 0, stream>>>(g);
  else if (amode == 0 && bmode == 1) gemm_bf16_kernel<0, 1><<<grid, 128, 0, stream>>>(g);
  else                               gemm_bf16_kernel<1, 1><<<grid, 128, 0, stream>>>(g);
}

// ---------------------------------------------------------------------------
// Streaming helpers
// ---------------------------------------------------------------------------

__global__ __launch_bounds__(256) void convert_f32_bf16(const float* __restrict__ in,
                                                        bf16_t* __restrict__ out,
                                                        long long n) {
  long long i = (long long)blockIdx.x * 256 + threadIdx.x;
  if (i < n) out[i] = (bf16_t)in[i];
}

__global__ __launch_bounds__(256) void fill_zero_f32(float* __restrict__ p, long long n) {
  long long i = (long long)blockIdx.x * 256 + threadIdx.x;
  if (i < n) p[i] = 0.0f;
}

// Bilinear downsample HxW -> 32x32, align_corners=False, antialias=False.
// For integer scale s in {2,4,8,16}: src = (o+0.5)*s-0.5 -> frac 0.5, i.e.
// the exact average of the 2x2 block starting at (s*o + s/2 - 1).
__global__ __launch_bounds__(256) void downsample_kernel(const float* __restrict__ in,
                                                         bf16_t* __restrict__ out,
                                                         int s, long long total) {
  long long i = (long long)blockIdx.x * 256 + threadIdx.x;
  if (i >= total) return;
  const int ox = (int)(i & 31);
  const int oy = (int)((i >> 5) & 31);
  const long long plane = i >> 10;  // b*C + c
  const int Hin = 32 * s;
  const long long base = plane * (long long)Hin * Hin;
  const int iy = s * oy + (s >> 1) - 1;
  const int ix = s * ox + (s >> 1) - 1;
  const float* p0 = in + base + (long long)iy * Hin + ix;
  __builtin_prefetch(p0 + 2 * Hin, 0, 1);  // -> global_prefetch_b8
  const float v = 0.25f * (p0[0] + p0[1] + p0[Hin] + p0[Hin + 1]);
  out[i] = (bf16_t)v;
}

// im2col (3x3, pad 1) on the 32x32 bf16 maps: col[p][k], p=b*1024+y*32+x,
// k = ci*9 + dy*3 + dx.
__global__ __launch_bounds__(256) void im2col_kernel(const bf16_t* __restrict__ xd,
                                                     bf16_t* __restrict__ col,
                                                     int Ci, long long total) {
  long long i = (long long)blockIdx.x * 256 + threadIdx.x;
  if (i >= total) return;
  const int K = Ci * 9;
  const long long p = i / K;
  const int k = (int)(i - p * K);
  const int ci = k / 9;
  const int rem = k - ci * 9;
  const int dy = rem / 3, dx = rem - dy * 3;
  const int b = (int)(p >> 10);
  const int y = (int)((p >> 5) & 31);
  const int x = (int)(p & 31);
  const int yy = y + dy - 1, xx = x + dx - 1;
  bf16_t v = (bf16_t)0.0f;
  if ((unsigned)yy < 32u && (unsigned)xx < 32u)
    v = xd[(((long long)b * Ci + ci) << 10) + yy * 32 + xx];
  col[i] = v;
}

__global__ __launch_bounds__(256) void bn_prep_kernel(const float* __restrict__ gam,
                                                      const float* __restrict__ bet,
                                                      const float* __restrict__ rm,
                                                      const float* __restrict__ rv,
                                                      float* __restrict__ scale,
                                                      float* __restrict__ bias) {
  int i = blockIdx.x * 256 + threadIdx.x;
  if (i < 512) {
    const float s = gam[i] / sqrtf(rv[i] + 1e-5f);
    scale[i] = s;
    bias[i]  = bet[i] - rm[i] * s;
  }
}

// posT[p][d] = rel_h[d, y] + rel_w[d, x]  (p = y*32 + x), d = 64.
__global__ __launch_bounds__(256) void build_posT_kernel(const float* __restrict__ relH,
                                                         const float* __restrict__ relW,
                                                         float* __restrict__ posT) {
  int i = blockIdx.x * 256 + threadIdx.x;  // 65536
  if (i >= 65536) return;
  const int d = i & 63;
  const int p = i >> 6;
  const int y = p >> 5, x = p & 31;
  posT[i] = relH[d * 32 + y] + relW[d * 32 + x];
}

// Row softmax over 1024 columns (f32 in -> bf16 out). One block per row.
__global__ __launch_bounds__(256) void softmax_kernel(const float* __restrict__ E,
                                                      bf16_t* __restrict__ att) {
  __shared__ float red[256];
  const int t = threadIdx.x;
  const long long row = blockIdx.x;
  const float* e = E + row * 1024;
  float v[4];
  float m = -3.4e38f;
#pragma unroll
  for (int j = 0; j < 4; ++j) { v[j] = e[t + j * 256]; m = fmaxf(m, v[j]); }
  red[t] = m; __syncthreads();
  for (int s = 128; s > 0; s >>= 1) {
    if (t < s) red[t] = fmaxf(red[t], red[t + s]);
    __syncthreads();
  }
  m = red[0]; __syncthreads();
  float sum = 0.0f;
#pragma unroll
  for (int j = 0; j < 4; ++j) { v[j] = expf(v[j] - m); sum += v[j]; }
  red[t] = sum; __syncthreads();
  for (int s = 128; s > 0; s >>= 1) {
    if (t < s) red[t] += red[t + s];
    __syncthreads();
  }
  const float inv = 1.0f / red[0];
#pragma unroll
  for (int j = 0; j < 4; ++j)
    att[row * 1024 + t + j * 256] = (bf16_t)(v[j] * inv);
}

// ---------------------------------------------------------------------------
// Host orchestration
// ---------------------------------------------------------------------------

static void run_attention(const bf16_t* srcQ, long long sQk, long long sQb,
                          const bf16_t* kv, long long kvB,
                          const float* resid,
                          const float* relH, const float* relW,
                          const float* wq, const float* bq,
                          const float* wk, const float* bk,
                          const float* wv, const float* bv,
                          const float* gamma,
                          float* outF, bf16_t* outH,
                          bf16_t* wqb, bf16_t* wkb, bf16_t* wvb,
                          float* posTf, bf16_t* Qtb, bf16_t* Ktb, bf16_t* Vb,
                          float* E, bf16_t* attb, hipStream_t stream) {
  convert_f32_bf16<<<(64 * 512) / 256, 256, 0, stream>>>(wq, wqb, 64 * 512);
  convert_f32_bf16<<<(64 * 512) / 256, 256, 0, stream>>>(wk, wkb, 64 * 512);
  convert_f32_bf16<<<(512 * 512) / 256, 256, 0, stream>>>(wv, wvb, 512 * 512);
  build_posT_kernel<<<65536 / 256, 256, 0, stream>>>(relH, relW, posTf);

  // Qt[b, n, d] = srcQ^T * Wq^T (+ bq)
  {
    GemmArgs g{};
    g.A = srcQ; g.sAm = 1;   g.sAk = sQk; g.aStrideB = sQb;
    g.B = wqb;  g.sBn = 512; g.sBk = 1;   g.bStrideB = 0;
    g.outH = Qtb; g.sCm = 64; g.cStrideB = 65536;
    g.biasN = bq;
    g.M = 1024; g.N = 64; g.K = 512; g.flags = F_BIASN | F_OUTH;
    launch_gemm(g, dim3(1, 16, 8), 1, 0, stream);
  }
  // Kt[b, n, d] = kv^T * Wk^T + bk + posT  (folds q.pos into q.(k+pos))
  {
    GemmArgs g{};
    g.A = kv;  g.sAm = 1;   g.sAk = 1024; g.aStrideB = kvB;
    g.B = wkb; g.sBn = 512; g.sBk = 1;    g.bStrideB = 0;
    g.outH = Ktb; g.sCm = 64; g.cStrideB = 65536;
    g.biasN = bk; g.addend = posTf; g.sDm = 64; g.dStrideB = 0;
    g.M = 1024; g.N = 64; g.K = 512; g.flags = F_BIASN | F_ADDEND | F_OUTH;
    launch_gemm(g, dim3(1, 16, 8), 1, 0, stream);
  }
  // V[b, c, n] = Wv * kv (+ bv)
  {
    GemmArgs g{};
    g.A = wvb; g.sAm = 512; g.sAk = 1;    g.aStrideB = 0;
    g.B = kv;  g.sBn = 1;   g.sBk = 1024; g.bStrideB = kvB;
    g.outH = Vb; g.sCm = 1024; g.cStrideB = 524288;
    g.biasM = bv;
    g.M = 512; g.N = 1024; g.K = 512; g.flags = F_BIASM | F_OUTH;
    launch_gemm(g, dim3(16, 8, 8), 0, 1, stream);
  }
  // E[b, n, m] = Qt * Kt^T
  {
    GemmArgs g{};
    g.A = Qtb; g.sAm = 64; g.sAk = 1; g.aStrideB = 65536;
    g.B = Ktb; g.sBn = 64; g.sBk = 1; g.bStrideB = 65536;
    g.outF = E; g.sCm = 1024; g.cStrideB = 1048576;
    g.M = 1024; g.N = 1024; g.K = 64; g.flags = F_OUTF;
    launch_gemm(g, dim3(16, 16, 8), 0, 0, stream);
  }
  softmax_kernel<<<8192, 256, 0, stream>>>(E, attb);
  // out[b, c, m] = gamma * (V * att^T) + residual
  {
    GemmArgs g{};
    g.A = Vb;   g.sAm = 1024; g.sAk = 1; g.aStrideB = 524288;
    g.B = attb; g.sBn = 1024; g.sBk = 1; g.bStrideB = 1048576;
    g.outF = outF; g.outH = outH; g.sCm = 1024; g.cStrideB = 524288;
    g.residual = resid; g.sRm = 1024; g.rStrideB = 524288;
    g.gammaPtr = gamma;
    g.M = 512; g.N = 1024; g.K = 1024;
    g.flags = F_GAMMARES | F_OUTF | (outH ? F_OUTH : 0);
    launch_gemm(g, dim3(16, 8, 8), 0, 0, stream);
  }
}

extern "C" void kernel_launch(void* const* d_in, const int* in_sizes, int n_in,
                              void* d_out, int out_size, void* d_ws, size_t ws_size,
                              hipStream_t stream) {
  (void)in_sizes; (void)n_in; (void)out_size; (void)ws_size;

  const float* x = (const float*)d_in[0];

  // --- workspace carve-out (256B aligned) ---
  char* W = (char*)d_ws;
  size_t off = 0;
  auto carve = [&](size_t bytes) -> char* {
    char* p = W + off;
    off += (bytes + 255) & ~(size_t)255;
    return p;
  };
  bf16_t* xb      = (bf16_t*)carve(8LL * 512 * 1024 * 2);       // x in bf16
  float*  multiX  = (float*) carve(512LL * 8192 * 4);           // conv accumulator
  bf16_t* multiXb = (bf16_t*)carve(512LL * 8192 * 2);
  bf16_t* xd      = (bf16_t*)carve(8LL * 256 * 1024 * 2);       // max-branch downsample
  char*   colE    = carve(37748736);                            // union: im2col / energy
  bf16_t* wconvb  = (bf16_t*)carve(512LL * 2304 * 2);
  float*  bnS     = (float*) carve(512 * 4);
  float*  bnB     = (float*) carve(512 * 4);
  bf16_t* wqb     = (bf16_t*)carve(64 * 512 * 2);
  bf16_t* wkb     = (bf16_t*)carve(64 * 512 * 2);
  bf16_t* wvb     = (bf16_t*)carve(512 * 512 * 2);
  float*  posTf   = (float*) carve(1024 * 64 * 4);
  bf16_t* Qtb     = (bf16_t*)carve(8LL * 1024 * 64 * 2);
  bf16_t* Ktb     = (bf16_t*)carve(8LL * 1024 * 64 * 2);
  bf16_t* Vb      = (bf16_t*)carve(8LL * 512 * 1024 * 2);
  bf16_t* attb    = (bf16_t*)carve(8LL * 1024 * 1024 * 2);
  float*  xenc    = (float*) carve(8LL * 512 * 1024 * 4);
  bf16_t* xencb   = (bf16_t*)carve(8LL * 512 * 1024 * 2);

  const long long NX = 8LL * 512 * 1024;  // 4,194,304
  convert_f32_bf16<<<(int)((NX + 255) / 256), 256, 0, stream>>>(x, xb, NX);
  fill_zero_f32<<<(int)((NX + 255) / 256), 256, 0, stream>>>(multiX, NX);

  // --- SPP branches: downsample -> im2col -> implicit-GEMM conv + BN + ReLU ---
  const int Ci[4] = {32, 64, 128, 256};
  const int sc[4] = {16, 8, 4, 2};
  for (int i = 0; i < 4; ++i) {
    const float* wI  = (const float*)d_in[5 + 5 * i];
    const float* gI  = (const float*)d_in[6 + 5 * i];
    const float* bI  = (const float*)d_in[7 + 5 * i];
    const float* rmI = (const float*)d_in[8 + 5 * i];
    const float* rvI = (const float*)d_in[9 + 5 * i];
    const int K = Ci[i] * 9;

    bn_prep_kernel<<<2, 256, 0, stream>>>(gI, bI, rmI, rvI, bnS, bnB);
    const long long nw = 512LL * K;
    convert_f32_bf16<<<(int)((nw + 255) / 256), 256, 0, stream>>>(wI, wconvb, nw);

    const long long nds = 8LL * Ci[i] * 1024;
    downsample_kernel<<<(int)((nds + 255) / 256), 256, 0, stream>>>(
        (const float*)d_in[1 + i], xd, sc[i], nds);

    const long long ncol = 8192LL * K;
    im2col_kernel<<<(int)((ncol + 255) / 256), 256, 0, stream>>>(
        xd, (bf16_t*)colE, Ci[i], ncol);

    GemmArgs g{};
    g.A = wconvb;        g.sAm = K; g.sAk = 1; g.aStrideB = 0;
    g.B = (bf16_t*)colE; g.sBn = K; g.sBk = 1; g.bStrideB = 0;
    g.outF = multiX; g.sCm = 8192; g.cStrideB = 0;
    g.scaleM = bnS; g.biasM = bnB;
    g.M = 512; g.N = 8192; g.K = K;
    g.flags = F_SCALEM | F_BIASM | F_RELU | F_ACCUM | F_OUTF;
    launch_gemm(g, dim3(128, 8, 1), 0, 0, stream);
  }
  convert_f32_bf16<<<(int)((NX + 255) / 256), 256, 0, stream>>>(multiX, multiXb, NX);

  // --- encoder: q from multi-scale, k/v from x, residual x ---
  run_attention(multiXb, /*sQk=*/8192, /*sQb=*/1024,
                xb, /*kvB=*/524288, x,
                (const float*)d_in[25], (const float*)d_in[26],
                (const float*)d_in[27], (const float*)d_in[28],
                (const float*)d_in[29], (const float*)d_in[30],
                (const float*)d_in[31], (const float*)d_in[32],
                (const float*)d_in[33],
                xenc, xencb,
                wqb, wkb, wvb, posTf, Qtb, Ktb, Vb, (float*)colE, attb, stream);

  // --- decoder: q from x, k/v from encoder output, residual x ---
  run_attention(xb, /*sQk=*/1024, /*sQb=*/524288,
                xencb, /*kvB=*/524288, x,
                (const float*)d_in[34], (const float*)d_in[35],
                (const float*)d_in[36], (const float*)d_in[37],
                (const float*)d_in[38], (const float*)d_in[39],
                (const float*)d_in[40], (const float*)d_in[41],
                (const float*)d_in[42],
                (float*)d_out, nullptr,
                wqb, wkb, wvb, posTf, Qtb, Ktb, Vb, (float*)colE, attb, stream);
}
